// CrossAttentionFusion_87797721465393
// MI455X (gfx1250) — matrix operand order repaired and checked by reading, not compiled, not run on hardware
//
#include <hip/hip_runtime.h>
#include <hip/hip_bf16.h>

typedef __attribute__((ext_vector_type(16))) __bf16 v16bf;
typedef __attribute__((ext_vector_type(8)))  float  v8f;
typedef int v4i_vec __attribute__((vector_size(16)));   // matches builtin prototype

__device__ __forceinline__ v8f wmma_bf16(v16bf a, v16bf b, v8f c) {
  return __builtin_amdgcn_wmma_f32_16x16x32_bf16(false, a, false, b, (short)0, c, false, false);
}

// ---- async global->LDS path (gfx1250), guarded so either toolchain compiles ----
#if defined(__has_builtin)
# if __has_builtin(__builtin_amdgcn_global_load_async_to_lds_b128)
#  define ASYNC_LDS 1
# endif
#endif

__device__ __forceinline__ void async_copy16(const void* g, void* l) {
#ifdef ASYNC_LDS
  __builtin_amdgcn_global_load_async_to_lds_b128(
      (__attribute__((address_space(1))) v4i_vec*)g,
      (__attribute__((address_space(3))) v4i_vec*)l, 0, 0);
#else
  (void)g; (void)l;
#endif
}
__device__ __forceinline__ void wait_async0() {
#ifdef ASYNC_LDS
# if __has_builtin(__builtin_amdgcn_s_wait_asynccnt)
  __builtin_amdgcn_s_wait_asynccnt(0);
# else
  asm volatile("s_wait_asynccnt 0" ::: "memory");
# endif
#endif
}

// ---------------------------------------------------------------------------
// Tiled GEMM, D[m][n] = sum_k A[m][k]*B[k][n] (+Res+bias if EPI), bf16 WMMA.
// TM=128 x TN=128 x TK=32, 256 threads = 8 waves; wave w owns rows [16w,16w+16).
// AK:  A k-contiguous (a_cs==1), else A m-contiguous (a_rs==1).
// BN:  B n-contiguous (b_cs==1), else B k-contiguous (b_rs==1).
// EPI: f32 output with fused residual + bias; else bf16 output. d_cs==1 always.
// ---------------------------------------------------------------------------
#define TM 128
#define TN 128
#define TK 32
#define SA 40    // TK+8 (bf16 elems per LDS row)
#define SB 136   // TN+8

template<bool AK, bool BN, bool EPI>
__global__ __launch_bounds__(256) void gemm_kernel(
    const float* __restrict__ A, long long a_rs, long long a_cs, long long a_bs,
    const float* __restrict__ Bm, long long b_rs, long long b_cs, long long b_bs,
    void* __restrict__ D, long long d_rs, long long d_bs,
    const float* __restrict__ Res, long long r_rs, long long r_cs, long long r_bs,
    const float* __restrict__ bias, int K)
{
  __shared__ __bf16 sA[TM][SA];   // [m][k]
  __shared__ __bf16 sB[TK][SB];   // [k][n]

  const int tid  = threadIdx.x;
  const int wave = tid >> 5, lane = tid & 31;
  const int hh   = lane >> 4, ml = lane & 15;
  const int n0   = blockIdx.x * TN, m0 = blockIdx.y * TM;
  const int bz   = blockIdx.z;
  const float* Ab = A  + (size_t)bz * a_bs;
  const float* Bb = Bm + (size_t)bz * b_bs;

  // per-thread staging bases (hoisted out of the k-loop; only += per step)
  const float* ap; __bf16* as; long long aQ, aKs;
  if (AK) {                       // float4 along k
    int r = tid >> 3, c = (tid & 7) * 4;
    ap = Ab + (size_t)(m0 + r) * a_rs + c;
    as = &sA[r][c];
    aQ = 32 * a_rs; aKs = TK;
  } else {                        // float4 along m, LDS-transposed store
    int mcc = (tid & 31) * 4, c = tid >> 5;
    ap = Ab + (m0 + mcc) + (size_t)c * a_cs;
    as = &sA[mcc][c];
    aQ = 8 * a_cs; aKs = (long long)TK * a_cs;
  }
  const float* bp; __bf16* bs; long long bQ, bKs;
  if (BN) {                       // float4 along n
    int r = tid >> 5, c = (tid & 31) * 4;
    bp = Bb + (size_t)r * b_rs + (n0 + c);
    bs = &sB[r][c];
    bQ = 8 * b_rs; bKs = (long long)TK * b_rs;
  } else {                        // float4 along k, LDS-transposed store
    int kcc = (tid & 7) * 4, c = tid >> 3;
    bp = Bb + kcc + (size_t)(n0 + c) * b_cs;
    bs = &sB[kcc][c];
    bQ = 32 * b_cs; bKs = TK;
  }

  v8f acc[8] = {};

  for (int k0 = 0; k0 < K; k0 += TK) {
    #pragma unroll
    for (int q2 = 0; q2 < 4; ++q2) {
      float4 fa = *reinterpret_cast<const float4*>(ap + q2 * aQ);
      if (AK) {
        __bf16* d = as + q2 * (32 * SA);
        d[0] = (__bf16)fa.x; d[1] = (__bf16)fa.y; d[2] = (__bf16)fa.z; d[3] = (__bf16)fa.w;
      } else {
        __bf16* d = as + q2 * 8;
        d[0 * SA] = (__bf16)fa.x; d[1 * SA] = (__bf16)fa.y;
        d[2 * SA] = (__bf16)fa.z; d[3 * SA] = (__bf16)fa.w;
      }
    }
    #pragma unroll
    for (int q2 = 0; q2 < 4; ++q2) {
      float4 fb = *reinterpret_cast<const float4*>(bp + q2 * bQ);
      if (BN) {
        __bf16* d = bs + q2 * (8 * SB);
        d[0] = (__bf16)fb.x; d[1] = (__bf16)fb.y; d[2] = (__bf16)fb.z; d[3] = (__bf16)fb.w;
      } else {
        __bf16* d = bs + q2 * 32;
        d[0 * SB] = (__bf16)fb.x; d[1 * SB] = (__bf16)fb.y;
        d[2 * SB] = (__bf16)fb.z; d[3 * SB] = (__bf16)fb.w;
      }
    }
    ap += aKs; bp += bKs;
    if (k0 + TK < K) { __builtin_prefetch(ap, 0, 0); __builtin_prefetch(bp, 0, 0); }
    __syncthreads();

    // A fragment: lane M = ml, K chunks at hh*8 and hh*8+16
    v16bf af;
    {
      const __bf16* afp = &sA[wave * 16 + ml][hh * 8];
      #pragma unroll
      for (int e = 0; e < 8; ++e) { af[e] = afp[e]; af[e + 8] = afp[e + 16]; }
    }
    // B fragments: lane N = ml, K chunks at hh*8 and hh*8+16
    #pragma unroll
    for (int sub = 0; sub < 8; ++sub) {
      v16bf bfr;
      const __bf16* bp2 = &sB[hh * 8][sub * 16 + ml];
      #pragma unroll
      for (int e = 0; e < 8; ++e) { bfr[e] = bp2[e * SB]; bfr[e + 8] = bp2[(e + 16) * SB]; }
      acc[sub] = wmma_bf16(af, bfr, acc[sub]);
    }
    __syncthreads();
  }

  // epilogue: D elem r -> M = r + 8*hh, N = ml + 16*sub
  #pragma unroll
  for (int sub = 0; sub < 8; ++sub) {
    #pragma unroll
    for (int r = 0; r < 8; ++r) {
      int gm = m0 + wave * 16 + r + hh * 8;
      int gn = n0 + sub * 16 + ml;
      float val = acc[sub][r];
      size_t di = (size_t)bz * d_bs + (size_t)gm * d_rs + gn;
      if (EPI) {
        val += Res[(size_t)bz * r_bs + (size_t)gm * r_rs + (size_t)gn * r_cs] + bias[gn];
        ((float*)D)[di] = val;
      } else {
        ((__bf16*)D)[di] = (__bf16)val;
      }
    }
  }
}

// ---------------------------------------------------------------------------
// Flash attention: q (B,L,C) bf16, kT (B,C,L) bf16, v (B,L,C) bf16 -> xo (B,L,C) f32
// Block = (64 query rows, one head, one batch). 4 waves, 16 rows each.
// ---------------------------------------------------------------------------
__global__ __launch_bounds__(128) void attn_kernel(
    const __bf16* __restrict__ q, const __bf16* __restrict__ kT,
    const __bf16* __restrict__ v, float* __restrict__ xo,
    int L, int C, float scale)
{
  __shared__ __bf16 sK[64][64 + 8];        // [d][m]
  __shared__ __bf16 sV[64][64 + 8];        // [m][d]
  __shared__ __bf16 sP[4][16][64 + 8];     // per-wave P tile [l][m]

  const int bz = blockIdx.z, h = blockIdx.y;
  const int l0 = blockIdx.x * 64;
  const int tid  = threadIdx.x;
  const int wave = tid >> 5, lane = tid & 31;
  const int hh = lane >> 4, ml = lane & 15;

  // Q fragments straight from global (contiguous along d)
  v16bf aq[2];
  {
    int lg = l0 + wave * 16 + ml;
    const __bf16* qp = q + ((size_t)bz * L + lg) * C + h * 64;
    #pragma unroll
    for (int ks = 0; ks < 2; ++ks) {
      const __bf16* p = qp + ks * 32 + hh * 8;
      #pragma unroll
      for (int e = 0; e < 8; ++e) { aq[ks][e] = p[e]; aq[ks][e + 8] = p[e + 16]; }
    }
  }

  float mrow[8], lrow[8];
  #pragma unroll
  for (int r = 0; r < 8; ++r) { mrow[r] = -1e30f; lrow[r] = 0.f; }
  v8f oacc[4] = {};

  for (int m0 = 0; m0 < L; m0 += 64) {
    // stage K tile [d][m] and V tile [m][d]; 16B per lane per issue
#ifdef ASYNC_LDS
    #pragma unroll
    for (int q2 = 0; q2 < 4; ++q2) {
      int ci = tid + 128 * q2;
      int rr = ci >> 3, ee = (ci & 7) * 8;
      async_copy16(kT + ((size_t)bz * C + h * 64 + rr) * L + m0 + ee, &sK[rr][ee]);
      async_copy16(v  + ((size_t)bz * L + m0 + rr) * C + h * 64 + ee, &sV[rr][ee]);
    }
    wait_async0();
#else
    for (int i = tid; i < 64 * 64; i += 128) {
      int d = i >> 6, m = i & 63;
      sK[d][m] = kT[((size_t)bz * C + h * 64 + d) * L + m0 + m];
    }
    for (int i = tid; i < 64 * 64; i += 128) {
      int m = i >> 6, d = i & 63;
      sV[m][d] = v[((size_t)bz * L + m0 + m) * C + h * 64 + d];
    }
#endif
    __syncthreads();

    // S = Q K^T  (16 x 64 per wave)
    v8f s[4] = {};
    #pragma unroll
    for (int sub = 0; sub < 4; ++sub) {
      #pragma unroll
      for (int ks = 0; ks < 2; ++ks) {
        v16bf bfr;
        const __bf16* p = &sK[ks * 32 + hh * 8][sub * 16 + ml];
        #pragma unroll
        for (int e = 0; e < 8; ++e) { bfr[e] = p[e * 72]; bfr[e + 8] = p[(e + 16) * 72]; }
        s[sub] = wmma_bf16(aq[ks], bfr, s[sub]);
      }
    }

    // streaming softmax (row M = r + 8*hh, col N = ml + 16*sub)
    float rmax[8];
    #pragma unroll
    for (int r = 0; r < 8; ++r) rmax[r] = -1e30f;
    #pragma unroll
    for (int sub = 0; sub < 4; ++sub)
      #pragma unroll
      for (int r = 0; r < 8; ++r) {
        s[sub][r] *= scale;
        rmax[r] = fmaxf(rmax[r], s[sub][r]);
      }
    #pragma unroll
    for (int off = 1; off < 16; off <<= 1)
      #pragma unroll
      for (int r = 0; r < 8; ++r) rmax[r] = fmaxf(rmax[r], __shfl_xor(rmax[r], off, 32));

    float corr[8], rsum[8];
    #pragma unroll
    for (int r = 0; r < 8; ++r) {
      float mnew = fmaxf(mrow[r], rmax[r]);
      corr[r] = __expf(mrow[r] - mnew);
      mrow[r] = mnew;
      rsum[r] = 0.f;
    }
    #pragma unroll
    for (int sub = 0; sub < 4; ++sub)
      #pragma unroll
      for (int r = 0; r < 8; ++r) {
        float p = __expf(s[sub][r] - mrow[r]);
        rsum[r] += p;
        sP[wave][r + hh * 8][ml + sub * 16] = (__bf16)p;
      }
    #pragma unroll
    for (int off = 1; off < 16; off <<= 1)
      #pragma unroll
      for (int r = 0; r < 8; ++r) rsum[r] += __shfl_xor(rsum[r], off, 32);
    #pragma unroll
    for (int r = 0; r < 8; ++r) lrow[r] = lrow[r] * corr[r] + rsum[r];
    #pragma unroll
    for (int sub = 0; sub < 4; ++sub)
      #pragma unroll
      for (int r = 0; r < 8; ++r) oacc[sub][r] *= corr[r];

    // O += P @ V  (same-wave LDS round trip; LDS in-order per wave)
    #pragma unroll
    for (int ks = 0; ks < 2; ++ks) {
      v16bf apf;
      const __bf16* pp = &sP[wave][ml][ks * 32 + hh * 8];
      #pragma unroll
      for (int e = 0; e < 8; ++e) { apf[e] = pp[e]; apf[e + 8] = pp[e + 16]; }
      #pragma unroll
      for (int sub = 0; sub < 4; ++sub) {
        v16bf bfr;
        const __bf16* vp = &sV[ks * 32 + hh * 8][sub * 16 + ml];
        #pragma unroll
        for (int e = 0; e < 8; ++e) { bfr[e] = vp[e * 72]; bfr[e + 8] = vp[(e + 16) * 72]; }
        oacc[sub] = wmma_bf16(apf, bfr, oacc[sub]);
      }
    }
    __syncthreads();
  }

  #pragma unroll
  for (int r = 0; r < 8; ++r) {
    float rinv = 1.0f / lrow[r];
    int lg = l0 + wave * 16 + r + hh * 8;
    #pragma unroll
    for (int sub = 0; sub < 4; ++sub) {
      int d = sub * 16 + ml;
      xo[((size_t)bz * L + lg) * C + h * 64 + d] = oacc[sub][r] * rinv;
    }
  }
}

// ---------------------------------------------------------------------------
// LayerNorm over C + transpose (B,L,C) -> (B,C,H,W). One wave per row.
// ---------------------------------------------------------------------------
__global__ __launch_bounds__(128) void ln_transpose_kernel(
    const float* __restrict__ z, const float* __restrict__ gamma,
    const float* __restrict__ beta, float* __restrict__ out,
    int L, int C, float eps)
{
  int row  = blockIdx.x * 4 + (threadIdx.x >> 5);   // over B*L
  int lane = threadIdx.x & 31;
  int bz = row / L, l = row % L;
  const float* zp = z + (size_t)row * C;

  float vals[16];
  float s = 0.f;
  #pragma unroll
  for (int i = 0; i < 16; ++i) { vals[i] = zp[lane + i * 32]; s += vals[i]; }
  #pragma unroll
  for (int off = 1; off < 32; off <<= 1) s += __shfl_xor(s, off, 32);
  float mu = s * (1.0f / 512.0f);

  float vs = 0.f;
  #pragma unroll
  for (int i = 0; i < 16; ++i) { float d = vals[i] - mu; vs += d * d; }
  #pragma unroll
  for (int off = 1; off < 32; off <<= 1) vs += __shfl_xor(vs, off, 32);
  float inv = rsqrtf(vs * (1.0f / 512.0f) + eps);

  #pragma unroll
  for (int i = 0; i < 16; ++i) {
    int c = lane + i * 32;
    out[((size_t)bz * C + c) * L + l] = (vals[i] - mu) * inv * gamma[c] + beta[c];
  }
}

// ---------------------------------------------------------------------------
extern "C" void kernel_launch(void* const* d_in, const int* in_sizes, int n_in,
                              void* d_out, int out_size, void* d_ws, size_t ws_size,
                              hipStream_t stream) {
  (void)in_sizes; (void)n_in; (void)out_size; (void)ws_size;
  const float* pet  = (const float*)d_in[0];
  const float* ct   = (const float*)d_in[1];
  const float* Wq   = (const float*)d_in[2];
  const float* Wk   = (const float*)d_in[3];
  const float* Wv   = (const float*)d_in[4];
  const float* Wp   = (const float*)d_in[5];
  const float* bp   = (const float*)d_in[6];
  const float* gamma= (const float*)d_in[7];
  const float* beta = (const float*)d_in[8];
  float* out = (float*)d_out;

  const int B = 2, C = 512, L = 4096, NH = 8;
  const long long CL = (long long)C * L;

  // workspace: q(bf16) | kT(bf16) | v(bf16) | x(f32); z(f32) reuses q+kT region
  size_t elems = (size_t)B * L * C;
  char* ws = (char*)d_ws;
  __bf16* qb = (__bf16*)(ws);
  __bf16* kb = (__bf16*)(ws + elems * 2);
  __bf16* vb = (__bf16*)(ws + elems * 4);
  float*  xw = (float*)(ws + elems * 6);
  float*  zw = (float*)(ws);            // dead q/kT space after attention

  dim3 blk(256);

  // kT (B,C,L) = Wk @ ct^T : A=Wk (k-contig), B=ct_feat (n-contig)
  gemm_kernel<true, true, false><<<dim3(L / TN, C / TM, B), blk, 0, stream>>>(
      Wk, C, 1, 0,  ct, L, 1, CL,
      kb, L, CL,
      nullptr, 0, 0, 0, nullptr, C);

  // q (B,L,C) = pet_seq @ Wq^T : A[m][k]=pet[k*L+m] (m-contig), B[k][n]=Wq[n*C+k] (k-contig)
  gemm_kernel<false, false, false><<<dim3(C / TN, L / TM, B), blk, 0, stream>>>(
      pet, 1, L, CL,  Wq, 1, C, 0,
      qb, C, CL,
      nullptr, 0, 0, 0, nullptr, C);

  // v (B,L,C) = ct_seq @ Wv^T
  gemm_kernel<false, false, false><<<dim3(C / TN, L / TM, B), blk, 0, stream>>>(
      ct, 1, L, CL,  Wv, 1, C, 0,
      vb, C, CL,
      nullptr, 0, 0, 0, nullptr, C);

  // flash attention -> xw (B,L,C) f32
  attn_kernel<<<dim3(L / 64, NH, B), dim3(128), 0, stream>>>(qb, kb, vb, xw, L, C, 0.125f);

  // z = pet_seq + x @ Wp^T + bp (f32, fused residual read of pet_feat^T)
  gemm_kernel<true, false, true><<<dim3(C / TN, L / TM, B), blk, 0, stream>>>(
      xw, C, 1, CL,  Wp, 1, C, 0,
      zw, C, CL,
      pet, 1, L, CL, bp, C);

  // LayerNorm + transpose to (B,C,H,W)
  ln_transpose_kernel<<<dim3((B * L) / 4), dim3(128), 0, stream>>>(zw, gamma, beta, out, L, C, 1e-5f);
}
